// HarmonicGAE_70463233458549
// MI455X (gfx1250) — matrix operand order repaired and checked by reading, not compiled, not run on hardware
//
#include <hip/hip_runtime.h>
#include <hip/hip_bf16.h>
#include <cstddef>

#define N_NODES 10000
#define N_EDGES 320000
#define NODE_DIM 24
#define HID 64

typedef __attribute__((ext_vector_type(16))) _Float16 v16h;
typedef __attribute__((ext_vector_type(8)))  _Float16 v8h;
typedef __attribute__((ext_vector_type(8)))  float    v8f;

// ---------------------------------------------------------------------------
// h = x @ W_in + b_in   (one thread per (node, hid))
// ---------------------------------------------------------------------------
__global__ __launch_bounds__(256) void node_embed_kernel(
    const float* __restrict__ x, const float* __restrict__ W_in,
    const float* __restrict__ b_in, float* __restrict__ h) {
  int idx = blockIdx.x * blockDim.x + threadIdx.x;
  if (idx >= N_NODES * HID) return;
  int i = idx >> 6, t = idx & 63;
  float acc = b_in[t];
  const float* xr = x + (size_t)i * NODE_DIM;
#pragma unroll
  for (int k = 0; k < NODE_DIM; ++k) acc += xr[k] * W_in[k * HID + t];
  h[idx] = acc;
}

// ---------------------------------------------------------------------------
// zero a float buffer
// ---------------------------------------------------------------------------
__global__ __launch_bounds__(256) void zero_kernel(float* __restrict__ p, int n) {
  int idx = blockIdx.x * blockDim.x + threadIdx.x;
  if (idx < n) p[idx] = 0.0f;
}

// ---------------------------------------------------------------------------
// Edge scatter: msg = relu(h[src] + (edge_attr @ W_e + b_e)); agg[dst] += msg
// 64 threads per edge; fp32 atomics into L2-resident agg (2.56 MB << 192 MB L2)
// ---------------------------------------------------------------------------
__global__ __launch_bounds__(256) void gine_scatter_kernel(
    const float* __restrict__ h, const float* __restrict__ edge_attr,
    const int* __restrict__ edge_index, const float* __restrict__ W_e,
    const float* __restrict__ b_e, float* __restrict__ agg) {
  int idx = blockIdx.x * blockDim.x + threadIdx.x;
  int e = idx >> 6;
  int t = idx & 63;
  if (e >= N_EDGES) return;
  int src = edge_index[e];            // row 0: source
  int dst = edge_index[N_EDGES + e];  // row 1: destination
  float a0 = edge_attr[(size_t)e * 2 + 0];
  float a1 = edge_attr[(size_t)e * 2 + 1];
  float ea = a0 * W_e[t] + a1 * W_e[HID + t] + b_e[t];
  float msg = h[(size_t)src * HID + t] + ea;
  msg = fmaxf(msg, 0.0f);
  atomicAdd(&agg[(size_t)dst * HID + t], msg);
}

// ---------------------------------------------------------------------------
// z = h + agg;  h = [relu]( relu(z @ W1 + b1) @ W2 + b2 )   (4 nodes / block)
// ---------------------------------------------------------------------------
__global__ __launch_bounds__(256) void gine_mlp_kernel(
    float* __restrict__ h, const float* __restrict__ agg,
    const float* __restrict__ W1, const float* __restrict__ b1,
    const float* __restrict__ W2, const float* __restrict__ b2,
    int apply_relu) {
  __shared__ float zs[4][HID];
  __shared__ float hm[4][HID];
  int local = threadIdx.x >> 6;  // 0..3
  int t = threadIdx.x & 63;
  int node = blockIdx.x * 4 + local;  // N_NODES % 4 == 0 -> always valid
  size_t off = (size_t)node * HID + t;
  zs[local][t] = h[off] + agg[off];
  __syncthreads();
  float acc = b1[t];
#pragma unroll 8
  for (int k = 0; k < HID; ++k) acc += zs[local][k] * W1[k * HID + t];
  hm[local][t] = fmaxf(acc, 0.0f);
  __syncthreads();
  float acc2 = b2[t];
#pragma unroll 8
  for (int k = 0; k < HID; ++k) acc2 += hm[local][k] * W2[k * HID + t];
  if (apply_relu) acc2 = fmaxf(acc2, 0.0f);
  h[off] = acc2;
}

// ---------------------------------------------------------------------------
// g = h @ M   (no bias)
// ---------------------------------------------------------------------------
__global__ __launch_bounds__(256) void matmul_hid_kernel(
    const float* __restrict__ h, const float* __restrict__ M,
    float* __restrict__ g) {
  int idx = blockIdx.x * blockDim.x + threadIdx.x;
  if (idx >= N_NODES * HID) return;
  int i = idx >> 6, t = idx & 63;
  const float* hr = h + (size_t)i * HID;
  float acc = 0.0f;
#pragma unroll 8
  for (int k = 0; k < HID; ++k) acc += hr[k] * M[k * HID + t];
  g[idx] = acc;
}

// ---------------------------------------------------------------------------
// fp32 -> (f16 hi, f16 lo residual) split for near-fp32-accurate f16 WMMA
// ---------------------------------------------------------------------------
__global__ __launch_bounds__(256) void split_f16_kernel(
    const float* __restrict__ src, _Float16* __restrict__ hi,
    _Float16* __restrict__ lo, int n) {
  int idx = blockIdx.x * blockDim.x + threadIdx.x;
  if (idx >= n) return;
  float v = src[idx];
  _Float16 vh = (_Float16)v;
  hi[idx] = vh;
  lo[idx] = (_Float16)(v - (float)vh);
}

// ---------------------------------------------------------------------------
// Decode: logits = g @ h^T via v_wmma_f32_16x16x32_f16 with hi/lo split.
// D = A x B + C: A = g tile (16 x 64), B = h^T tile (64 x 16).
// k order used here (contiguous, same for A and B; not AMD's table order):
//   A lane (m = lane&15, half = lane>>4): element p of v16h holds
//     A[m][k0 + half*16 + p]       for p in 0..15     (two contiguous v8h loads)
//   B lane (n = lane&15, half): element p holds B[k0 + half*16 + p][n]
//     = h[jBase+n][k0 + half*16 + p]                  (one contiguous v16h load)
// x = hi + lo  =>  A*B ~= Ah*Bh + Ah*Bl + Al*Bh  (drop lo*lo): 6 WMMAs per tile.
// One wave per 16-row i-tile; j-tiles streamed (h/g tiles are L2-resident).
// ---------------------------------------------------------------------------
__global__ __launch_bounds__(256) void decode_wmma_kernel(
    const _Float16* __restrict__ gh, const _Float16* __restrict__ gl,
    const _Float16* __restrict__ hh, const _Float16* __restrict__ hl,
    float* __restrict__ out) {
  const int lane = threadIdx.x & 31;
  const int wave = threadIdx.x >> 5;
  const int iTile = blockIdx.x * 8 + wave;  // 16-row tile index (625 total)
  if (iTile >= N_NODES / 16) return;        // wave-uniform; EXEC stays all-1s
  const int iBase = iTile * 16;
  const int half = lane >> 4;   // 0 or 1
  const int mn = lane & 15;
  const int half8 = half * 8;

  // Preload A (g rows iBase..iBase+15), both K blocks, hi and lo.
  const _Float16* gArow_h = gh + (size_t)(iBase + mn) * HID;
  const _Float16* gArow_l = gl + (size_t)(iBase + mn) * HID;
  v16h aH[2], aL[2];
#pragma unroll
  for (int kk = 0; kk < 2; ++kk) {
    const int k0 = kk * 32;
    v8h h0 = *(const v8h*)(gArow_h + k0 + 2 * half8);
    v8h h1 = *(const v8h*)(gArow_h + k0 + 8 + 2 * half8);
    aH[kk] = __builtin_shufflevector(h0, h1, 0, 1, 2, 3, 4, 5, 6, 7, 8, 9, 10,
                                     11, 12, 13, 14, 15);
    v8h l0 = *(const v8h*)(gArow_l + k0 + 2 * half8);
    v8h l1 = *(const v8h*)(gArow_l + k0 + 8 + 2 * half8);
    aL[kk] = __builtin_shufflevector(l0, l1, 0, 1, 2, 3, 4, 5, 6, 7, 8, 9, 10,
                                     11, 12, 13, 14, 15);
  }

  // j-tile range for this gridDim.y chunk
  const int JT = N_NODES / 16;  // 625
  const int chunk = 63;         // 10 * 63 >= 625
  int jt0 = blockIdx.y * chunk;
  int jt1 = jt0 + chunk;
  if (jt1 > JT) jt1 = JT;

  for (int jt = jt0; jt < jt1; ++jt) {
    const int jBase = jt * 16;
    const _Float16* bh = hh + (size_t)(jBase + mn) * HID + half * 16;
    const _Float16* bl = hl + (size_t)(jBase + mn) * HID + half * 16;
    if (jt + 1 < jt1) {  // prefetch next B tile (global_prefetch_b8 path)
      __builtin_prefetch((const void*)(bh + 16 * HID), 0, 1);
      __builtin_prefetch((const void*)(bl + 16 * HID), 0, 1);
    }
    v16h bH0 = *(const v16h*)(bh);
    v16h bH1 = *(const v16h*)(bh + 32);
    v16h bL0 = *(const v16h*)(bl);
    v16h bL1 = *(const v16h*)(bl + 32);

    v8f acc = {};
    acc = __builtin_amdgcn_wmma_f32_16x16x32_f16(false, aH[0], false, bH0,
                                                 (short)0, acc, false, false);
    acc = __builtin_amdgcn_wmma_f32_16x16x32_f16(false, aH[1], false, bH1,
                                                 (short)0, acc, false, false);
    acc = __builtin_amdgcn_wmma_f32_16x16x32_f16(false, aH[0], false, bL0,
                                                 (short)0, acc, false, false);
    acc = __builtin_amdgcn_wmma_f32_16x16x32_f16(false, aH[1], false, bL1,
                                                 (short)0, acc, false, false);
    acc = __builtin_amdgcn_wmma_f32_16x16x32_f16(false, aL[0], false, bH0,
                                                 (short)0, acc, false, false);
    acc = __builtin_amdgcn_wmma_f32_16x16x32_f16(false, aL[1], false, bH1,
                                                 (short)0, acc, false, false);

    // D layout: VGPR r, lanes 0-15 -> row r; lanes 16-31 -> row r+8
#pragma unroll
    for (int r = 0; r < 8; ++r) {
      int row = iBase + r + half * 8;
      out[(size_t)row * N_NODES + jBase + mn] = acc[r];
    }
  }
}

// ---------------------------------------------------------------------------
// Row-wise online softmax stats: rmax[i], rsum[i] = sum exp(x - rmax)
// ---------------------------------------------------------------------------
__global__ __launch_bounds__(256) void row_stats_kernel(
    const float* __restrict__ logits, float* __restrict__ rmax,
    float* __restrict__ rsum) {
  __shared__ float sm[256];
  __shared__ float ss[256];
  int row = blockIdx.x;
  int tid = threadIdx.x;
  const float* p = logits + (size_t)row * N_NODES;
  float m = -3.4e38f, s = 0.0f;
  for (int j = tid; j < N_NODES; j += 256) {
    float v = p[j];
    if (v > m) {
      s = s * __expf(m - v) + 1.0f;
      m = v;
    } else {
      s += __expf(v - m);
    }
  }
  sm[tid] = m;
  ss[tid] = s;
  __syncthreads();
  for (int off = 128; off > 0; off >>= 1) {
    if (tid < off) {
      float m2 = sm[tid + off], s2 = ss[tid + off];
      float m1 = sm[tid], s1 = ss[tid];
      float mn = fmaxf(m1, m2);
      sm[tid] = mn;
      ss[tid] = s1 * __expf(m1 - mn) + s2 * __expf(m2 - mn);
    }
    __syncthreads();
  }
  if (tid == 0) {
    rmax[row] = sm[0];
    rsum[row] = ss[0];
  }
}

// ---------------------------------------------------------------------------
// In-place normalize: out = exp(x - rmax) / rsum
// ---------------------------------------------------------------------------
__global__ __launch_bounds__(256) void row_norm_kernel(
    float* __restrict__ out, const float* __restrict__ rmax,
    const float* __restrict__ rsum) {
  int row = blockIdx.x;
  float mx = rmax[row];
  float inv = 1.0f / rsum[row];
  float* p = out + (size_t)row * N_NODES;
  for (int j = threadIdx.x; j < N_NODES; j += 256) {
    p[j] = __expf(p[j] - mx) * inv;
  }
}

// ---------------------------------------------------------------------------
extern "C" void kernel_launch(void* const* d_in, const int* in_sizes, int n_in,
                              void* d_out, int out_size, void* d_ws,
                              size_t ws_size, hipStream_t stream) {
  (void)in_sizes; (void)n_in; (void)out_size; (void)ws_size;
  // setup_inputs() dict order:
  const float* x         = (const float*)d_in[0];
  const float* edge_attr = (const float*)d_in[1];
  const int*   edge_idx  = (const int*)d_in[2];
  const float* W_in      = (const float*)d_in[3];
  const float* b_in      = (const float*)d_in[4];
  const float* W_e       = (const float*)d_in[5];
  const float* b_e       = (const float*)d_in[6];
  const float* W1        = (const float*)d_in[7];
  const float* b1        = (const float*)d_in[8];
  const float* W2        = (const float*)d_in[9];
  const float* b2        = (const float*)d_in[10];
  const float* M         = (const float*)d_in[11];
  float* out = (float*)d_out;

  const int NH = N_NODES * HID;  // 640000
  float* ws   = (float*)d_ws;
  float* h    = ws;              // [N, HID]
  float* agg  = ws + NH;         // [N, HID]
  float* g    = ws + 2 * NH;     // [N, HID]
  float* rmax = ws + 3 * NH;     // [N]
  float* rsum = rmax + N_NODES;  // [N]
  _Float16* hh = (_Float16*)(rsum + N_NODES);
  _Float16* hl = hh + NH;
  _Float16* gh = hl + NH;
  _Float16* gl = gh + NH;

  const int elemBlocks = (NH + 255) / 256;  // 2500

  // h0 = x @ W_in + b_in
  node_embed_kernel<<<elemBlocks, 256, 0, stream>>>(x, W_in, b_in, h);

  // Two GINE layers (shared MLP params); outer relu only after layer 1
  for (int layer = 0; layer < 2; ++layer) {
    zero_kernel<<<elemBlocks, 256, 0, stream>>>(agg, NH);
    gine_scatter_kernel<<<(N_EDGES * HID) / 256, 256, 0, stream>>>(
        h, edge_attr, edge_idx, W_e, b_e, agg);
    gine_mlp_kernel<<<N_NODES / 4, 256, 0, stream>>>(h, agg, W1, b1, W2, b2,
                                                     layer == 0 ? 1 : 0);
  }

  // g = h @ M
  matmul_hid_kernel<<<elemBlocks, 256, 0, stream>>>(h, M, g);

  // f16 hi/lo split of h and g for WMMA decode
  split_f16_kernel<<<elemBlocks, 256, 0, stream>>>(h, hh, hl, NH);
  split_f16_kernel<<<elemBlocks, 256, 0, stream>>>(g, gh, gl, NH);

  // logits = g @ h^T  (raw, into d_out)
  dim3 dgrid((N_NODES / 16 + 7) / 8, 10);  // 79 x 10 blocks, 8 waves each
  decode_wmma_kernel<<<dgrid, 256, 0, stream>>>(gh, gl, hh, hl, out);

  // softmax over axis 1
  row_stats_kernel<<<N_NODES, 256, 0, stream>>>(out, rmax, rsum);
  row_norm_kernel<<<N_NODES, 256, 0, stream>>>(out, rmax, rsum);
}
